// MDNRNNDynamics_80573586473616
// MI455X (gfx1250) — matrix operand order repaired and checked by reading, not compiled, not run on hardware
//
#include <hip/hip_runtime.h>
#include <cstdint>
#include <cstddef>

// Problem constants
#define BDIM 256
#define TSTEPS 1024
#define LDIM 32
#define ADIM 8
#define HDIM 256
#define KMIX 5
#define HEADN_REAL 327              // 5 + 160 + 160 + 1 + 1
#define HEADN_PAD 336               // 21 tiles of 16
#define NT_HEAD 21
#define GATES 1024                  // 4*H

typedef __attribute__((ext_vector_type(16))) __bf16 bf16x16;
typedef __attribute__((ext_vector_type(8)))  __bf16 bf16x8;
typedef __attribute__((ext_vector_type(8)))  float  floatx8;

// ---------- helpers ----------
__device__ __forceinline__ __bf16 f2bf(float f) {
  union { float f; unsigned int u; } in; in.f = f;
  unsigned int u = in.u;
  unsigned short s = (unsigned short)((u + 0x7FFFu + ((u >> 16) & 1u)) >> 16);
  union { unsigned short s; __bf16 b; } out; out.s = s;
  return out.b;
}

__device__ __forceinline__ float sigmoidf(float x) {
  return 1.0f / (1.0f + __expf(-x));
}

__device__ __forceinline__ unsigned int hash_u32(unsigned int x) {
  x ^= x >> 17; x *= 0xed5ad4bbu;
  x ^= x >> 11; x *= 0xac4c1b51u;
  x ^= x >> 15; x *= 0x31848babu;
  x ^= x >> 14;
  return x;
}
__device__ __forceinline__ float u01(unsigned int h) {
  return ((float)(h >> 8) + 0.5f) * (1.0f / 16777216.0f);  // strictly in (0,1)
}

// Build an A fragment (16x32 bf16) from a row-major bf16 LDS array.
// Layout used here (contiguous K, not the ISA table order): lanes 0-15 -> M=lane,
// elems 0..15 = K 0..15; lanes 16-31 hold K 16..31 (matches B packing).
template <int STRIDE>
__device__ __forceinline__ bf16x16 load_a_frag(const __bf16* base, int arow,
                                               int k0, int khalf) {
  const __bf16* p = base + arow * STRIDE + k0 + khalf * 16;
  bf16x8 lo = *reinterpret_cast<const bf16x8*>(p);
  bf16x8 hi = *reinterpret_cast<const bf16x8*>(p + 8);
  bf16x16 a;
#pragma unroll
  for (int j = 0; j < 8; ++j) { a[j] = lo[j]; a[j + 8] = hi[j]; }
  return a;
}

// ---------- weight pre-swizzle: fp32 -> bf16 WMMA B-fragments ----------
// Fragment element e: j=e&15, lane=(e>>4)&31, frag=e>>9.
// B[k][n] = W[n][k] with n = tn*16 + (lane&15), k = tk*32 + (lane>>4)*16 + j.
#define WIH_KT 2   // K = 40 padded to 64
#define WHH_KT 8   // K = 256
#define WHD_KT 8   // K = 256
#define WIH_ELEMS (64 * WIH_KT * 512)
#define WHH_ELEMS (64 * WHH_KT * 512)
#define WHD_ELEMS (NT_HEAD * WHD_KT * 512)

__global__ void mdnrnn_convert_weights(
    const float* __restrict__ W_ih, const float* __restrict__ b_ih,
    const float* __restrict__ W_hh, const float* __restrict__ b_hh,
    const float* __restrict__ pi_W, const float* __restrict__ pi_b,
    const float* __restrict__ mu_W, const float* __restrict__ mu_b,
    const float* __restrict__ lv_W, const float* __restrict__ lv_b,
    const float* __restrict__ r_W,  const float* __restrict__ r_b,
    const float* __restrict__ d_W,  const float* __restrict__ d_b,
    float* __restrict__ bias_g, float* __restrict__ bias_head,
    __bf16* __restrict__ wih_f, __bf16* __restrict__ whh_f,
    __bf16* __restrict__ whd_f) {
  int gid = blockIdx.x * blockDim.x + threadIdx.x;

  if (gid < GATES) { bias_g[gid] = b_ih[gid] + b_hh[gid]; return; }
  gid -= GATES;

  if (gid < HEADN_PAD) {
    int n = gid; float v = 0.0f;
    if (n < 5)        v = pi_b[n];
    else if (n < 165) v = mu_b[n - 5];
    else if (n < 325) v = lv_b[n - 165];
    else if (n == 325) v = r_b[0];
    else if (n == 326) v = d_b[0];
    bias_head[n] = v; return;
  }
  gid -= HEADN_PAD;

  if (gid < WIH_ELEMS) {
    int e = gid, j = e & 15, lane = (e >> 4) & 31, frag = e >> 9;
    int tk = frag % WIH_KT, tn = frag / WIH_KT;
    int n = tn * 16 + (lane & 15);
    int k = tk * 32 + (lane >> 4) * 16 + j;
    wih_f[e] = f2bf((k < (LDIM + ADIM)) ? W_ih[n * (LDIM + ADIM) + k] : 0.0f);
    return;
  }
  gid -= WIH_ELEMS;

  if (gid < WHH_ELEMS) {
    int e = gid, j = e & 15, lane = (e >> 4) & 31, frag = e >> 9;
    int tk = frag % WHH_KT, tn = frag / WHH_KT;
    int n = tn * 16 + (lane & 15);
    int k = tk * 32 + (lane >> 4) * 16 + j;
    whh_f[e] = f2bf(W_hh[n * HDIM + k]);
    return;
  }
  gid -= WHH_ELEMS;

  if (gid < WHD_ELEMS) {
    int e = gid, j = e & 15, lane = (e >> 4) & 31, frag = e >> 9;
    int tk = frag % WHD_KT, tn = frag / WHD_KT;
    int n = tn * 16 + (lane & 15);
    int k = tk * 32 + (lane >> 4) * 16 + j;
    float v = 0.0f;
    if (n < 5)         v = pi_W[n * HDIM + k];
    else if (n < 165)  v = mu_W[(n - 5) * HDIM + k];
    else if (n < 325)  v = lv_W[(n - 165) * HDIM + k];
    else if (n == 325) v = r_W[k];
    else if (n == 326) v = d_W[k];
    whd_f[e] = f2bf(v);
  }
}

// ---------- persistent recurrent kernel: 16 blocks x 16 batch rows ----------
// LDS carve (dynamic):
//   gates f32 [16][1024]  65536 B
//   h     f32 [16][256]   16384 B
//   c     f32 [16][256]   16384 B
//   head  f32 [16][336]   21504 B
//   h16  bf16 [16][256]    8192 B
//   x16  bf16 [16][64]     2048 B     total 130048 B
#define SMEM_BYTES 130048

__global__ __launch_bounds__(256, 1) void mdnrnn_persistent(
    const float* __restrict__ latent, const float* __restrict__ action,
    const int* __restrict__ dones,
    const float* __restrict__ bias_g, const float* __restrict__ bias_head,
    const __bf16* __restrict__ wih_f, const __bf16* __restrict__ whh_f,
    const __bf16* __restrict__ whd_f, float* __restrict__ out) {
  extern __shared__ char smem[];
  float*  sh_gates = (float*)(smem);
  float*  sh_h     = (float*)(smem + 65536);
  float*  sh_c     = (float*)(smem + 81920);
  float*  sh_head  = (float*)(smem + 98304);
  __bf16* sh_h16   = (__bf16*)(smem + 119808);
  __bf16* sh_x16   = (__bf16*)(smem + 128000);

  const int tid   = threadIdx.x;
  const int wave  = tid >> 5;
  const int lane  = tid & 31;
  const int khalf = lane >> 4;
  const int col   = lane & 15;   // N column within a 16x16 tile (B/C/D layout)
  const int arow  = lane & 15;   // M row of A fragment
  const int b0    = blockIdx.x * 16;

  const size_t BT = (size_t)BDIM * TSTEPS;
  float* out_nl = out;
  float* out_pi = out + BT * LDIM;
  float* out_mu = out_pi + BT * KMIX;
  float* out_lv = out_mu + BT * (KMIX * LDIM);
  float* out_r  = out_lv + BT * (KMIX * LDIM);
  float* out_d  = out_r + BT;

  // Loop-invariant bias values -> registers (one per owned N-tile).
  float gbias[8];
#pragma unroll
  for (int nt = 0; nt < 8; ++nt)
    gbias[nt] = bias_g[(wave * 8 + nt) * 16 + col];
  float hbias[3];
#pragma unroll
  for (int i = 0; i < 3; ++i) {
    int tn = wave + i * 8;
    hbias[i] = (tn < NT_HEAD) ? bias_head[tn * 16 + col] : 0.0f;
  }

  // init hidden state
  for (int i = tid; i < 16 * HDIM; i += 256) { sh_h[i] = 0.0f; sh_c[i] = 0.0f; }
  __syncthreads();

  for (int t = 0; t < TSTEPS; ++t) {
    // ---- 1. mask h/c by prev_done, refresh bf16 h, stage x_t in LDS ----
    for (int i = tid; i < 16 * HDIM; i += 256) {
      int row = i >> 8;
      int pd = (t == 0) ? 0 : dones[(size_t)(b0 + row) * TSTEPS + (t - 1)];
      float hv = sh_h[i];
      if (pd) { hv = 0.0f; sh_h[i] = 0.0f; sh_c[i] = 0.0f; }
      sh_h16[i] = f2bf(hv);
    }
    for (int i = tid; i < 16 * 64; i += 256) {
      int row = i >> 6, k = i & 63;
      float v = 0.0f;
      size_t bt = (size_t)(b0 + row) * TSTEPS + t;
      if (k < LDIM)             v = latent[bt * LDIM + k];
      else if (k < LDIM + ADIM) v = action[bt * ADIM + (k - LDIM)];
      sh_x16[i] = f2bf(v);
    }
    __syncthreads();

    // ---- 2. gates = x @ Wih^T + h @ Whh^T + bias   (WMMA bf16 -> f32) ----
    // A fragments are shared across the 8 owned N-tiles -> registers.
    // Partial unroll (2) keeps only ~2 tiles of B-fragments in flight:
    // enough to double-buffer global loads against WMMA chains without
    // spilling to scratch.
    {
      bf16x16 ax[WIH_KT], ah[WHH_KT];
#pragma unroll
      for (int tk = 0; tk < WIH_KT; ++tk)
        ax[tk] = load_a_frag<64>(sh_x16, arow, tk * 32, khalf);
#pragma unroll
      for (int tk = 0; tk < WHH_KT; ++tk)
        ah[tk] = load_a_frag<HDIM>(sh_h16, arow, tk * 32, khalf);

#pragma unroll 2
      for (int nt = 0; nt < 8; ++nt) {
        int tn = wave * 8 + nt;                 // 64 N-tiles across 8 waves
        float bv = gbias[nt];
        floatx8 acc = {bv, bv, bv, bv, bv, bv, bv, bv};
#pragma unroll
        for (int tk = 0; tk < WIH_KT; ++tk) {
          bf16x16 bm = reinterpret_cast<const bf16x16*>(
              wih_f + (size_t)(tn * WIH_KT + tk) * 512)[lane];
          acc = __builtin_amdgcn_wmma_f32_16x16x32_bf16(
              false, ax[tk], false, bm, (short)0, acc, false, false);
        }
#pragma unroll
        for (int tk = 0; tk < WHH_KT; ++tk) {
          bf16x16 bm = reinterpret_cast<const bf16x16*>(
              whh_f + (size_t)(tn * WHH_KT + tk) * 512)[lane];
          acc = __builtin_amdgcn_wmma_f32_16x16x32_bf16(
              false, ah[tk], false, bm, (short)0, acc, false, false);
        }
#pragma unroll
        for (int v = 0; v < 8; ++v)
          sh_gates[(v + khalf * 8) * GATES + tn * 16 + col] = acc[v];
      }
    }
    __syncthreads();

    // ---- 3. LSTM cell update ----
    for (int i = tid; i < 16 * HDIM; i += 256) {
      int row = i >> 8, u = i & 255;
      const float* g = sh_gates + row * GATES;
      float ig = g[u], fg = g[HDIM + u], gg = g[2 * HDIM + u], og = g[3 * HDIM + u];
      float c = sigmoidf(fg) * sh_c[i] + sigmoidf(ig) * tanhf(gg);
      float h = sigmoidf(og) * tanhf(c);
      sh_c[i] = c; sh_h[i] = h; sh_h16[i] = f2bf(h);
    }
    __syncthreads();

    // ---- 4. heads = h @ Whead^T + bias  (21 N-tiles over 8 waves) ----
    {
      bf16x16 ah[WHD_KT];
#pragma unroll
      for (int tk = 0; tk < WHD_KT; ++tk)
        ah[tk] = load_a_frag<HDIM>(sh_h16, arow, tk * 32, khalf);

#pragma unroll 1
      for (int it = 0; it < 3; ++it) {
        int tn = wave + it * 8;
        if (tn >= NT_HEAD) break;               // uniform per wave
        float bv = hbias[it];
        floatx8 acc = {bv, bv, bv, bv, bv, bv, bv, bv};
#pragma unroll
        for (int tk = 0; tk < WHD_KT; ++tk) {
          bf16x16 bm = reinterpret_cast<const bf16x16*>(
              whd_f + (size_t)(tn * WHD_KT + tk) * 512)[lane];
          acc = __builtin_amdgcn_wmma_f32_16x16x32_bf16(
              false, ah[tk], false, bm, (short)0, acc, false, false);
        }
#pragma unroll
        for (int v = 0; v < 8; ++v)
          sh_head[(v + khalf * 8) * HEADN_PAD + tn * 16 + col] = acc[v];
      }
    }
    __syncthreads();

    // ---- 5. write head outputs + sample next_latent ----
    for (int e = tid; e < 16 * HEADN_REAL; e += 256) {
      int row = e / HEADN_REAL, j = e % HEADN_REAL;
      float v = sh_head[row * HEADN_PAD + j];
      size_t bt = (size_t)(b0 + row) * TSTEPS + t;
      if (j < 5)         out_pi[bt * KMIX + j] = v;
      else if (j < 165)  out_mu[bt * 160 + (j - 5)] = v;
      else if (j < 325)  out_lv[bt * 160 + (j - 165)] = v;
      else if (j == 325) out_r[bt] = v;
      else               out_d[bt] = v;
    }
    for (int e = tid; e < 16 * LDIM; e += 256) {
      int row = e >> 5, dim = e & 31;
      int b = b0 + row;
      unsigned int seed = (unsigned int)(b * TSTEPS + t);
      // Gumbel-max categorical over pi_logits (temperature = 1)
      int idx = 0; float best = -3.4e38f;
#pragma unroll
      for (int k = 0; k < KMIX; ++k) {
        float u = u01(hash_u32(seed * 6271u + (unsigned)k * 2654435761u + 0x9e3779b9u));
        float gv = sh_head[row * HEADN_PAD + k] - __logf(-__logf(u));
        if (gv > best) { best = gv; idx = k; }
      }
      float mu = sh_head[row * HEADN_PAD + 5 + idx * LDIM + dim];
      float lv = sh_head[row * HEADN_PAD + 165 + idx * LDIM + dim];
      // Box-Muller normal
      unsigned int s2 = hash_u32(seed ^ 0xA511E9B3u) + (unsigned)dim * 0x85EBCA77u;
      float u1 = u01(hash_u32(s2 * 2246822519u + 374761393u));
      float u2 = u01(hash_u32(s2 * 3266489917u + 668265263u));
      float z = __fsqrt_rn(-2.0f * __logf(u1)) * __cosf(6.28318530718f * u2);
      out_nl[((size_t)b * TSTEPS + t) * LDIM + dim] = mu + __expf(0.5f * lv) * z;
    }
    __syncthreads();
  }
}

// ---------- launch ----------
extern "C" void kernel_launch(void* const* d_in, const int* in_sizes, int n_in,
                              void* d_out, int out_size, void* d_ws, size_t ws_size,
                              hipStream_t stream) {
  (void)in_sizes; (void)n_in; (void)out_size; (void)ws_size;
  const float* latent = (const float*)d_in[0];
  const float* action = (const float*)d_in[1];
  const int*   dones  = (const int*)d_in[2];
  const float* W_ih = (const float*)d_in[3];
  const float* b_ih = (const float*)d_in[4];
  const float* W_hh = (const float*)d_in[5];
  const float* b_hh = (const float*)d_in[6];
  const float* pi_W = (const float*)d_in[7];
  const float* pi_b = (const float*)d_in[8];
  const float* mu_W = (const float*)d_in[9];
  const float* mu_b = (const float*)d_in[10];
  const float* lv_W = (const float*)d_in[11];
  const float* lv_b = (const float*)d_in[12];
  const float* r_W  = (const float*)d_in[13];
  const float* r_b  = (const float*)d_in[14];
  const float* d_W  = (const float*)d_in[15];
  const float* d_b  = (const float*)d_in[16];

  // workspace carve (all 32B aligned)
  char* ws = (char*)d_ws;
  float*  bias_g    = (float*)(ws);                               // 4096 B
  float*  bias_head = (float*)(ws + 4096);                        // 1344 B
  __bf16* wih_f     = (__bf16*)(ws + 5440);                       // 131072 B
  __bf16* whh_f     = (__bf16*)(ws + 5440 + 131072);              // 524288 B
  __bf16* whd_f     = (__bf16*)(ws + 5440 + 131072 + 524288);     // 172032 B

  const int conv_items = GATES + HEADN_PAD + WIH_ELEMS + WHH_ELEMS + WHD_ELEMS;
  const int conv_blocks = (conv_items + 255) / 256;
  mdnrnn_convert_weights<<<conv_blocks, 256, 0, stream>>>(
      W_ih, b_ih, W_hh, b_hh, pi_W, pi_b, mu_W, mu_b, lv_W, lv_b,
      r_W, r_b, d_W, d_b, bias_g, bias_head, wih_f, whh_f, whd_f);

  (void)hipFuncSetAttribute((const void*)mdnrnn_persistent,
                            hipFuncAttributeMaxDynamicSharedMemorySize,
                            SMEM_BYTES);
  mdnrnn_persistent<<<16, 256, SMEM_BYTES, stream>>>(
      latent, action, dones, bias_g, bias_head, wih_f, whh_f, whd_f,
      (float*)d_out);
}
